// GCN_41618233098634
// MI455X (gfx1250) — matrix operand order repaired and checked by reading, not compiled, not run on hardware
//
#include <hip/hip_runtime.h>

#define N_NODES  100000
#define N_EDGES  3200000
#define D        128
#define N_GRAPHS 512

typedef __attribute__((ext_vector_type(16))) __bf16    v16bf;
typedef __attribute__((ext_vector_type(8)))  float     v8f;
typedef __attribute__((ext_vector_type(4)))  unsigned  v4u;
typedef __attribute__((ext_vector_type(8)))  unsigned  v8u;

__device__ __forceinline__ __bf16 f2bf(float f) {
    unsigned u = __builtin_bit_cast(unsigned, f);
    unsigned r = (u + 0x7FFFu + ((u >> 16) & 1u)) >> 16;
    unsigned short s = (unsigned short)r;
    return __builtin_bit_cast(__bf16, s);
}

// ---------------- utility ----------------
__global__ void k_zero(float* p, int n) {
    int i = blockIdx.x * blockDim.x + threadIdx.x;
    if (i < n) p[i] = 0.f;
}

// Convert W1/W2/W3 (fp32, K-major 128x128) into fragment-ordered bf16:
// wb[l*16384 + ((wcol*4 + kstep)*32 + lane)*16 + i]  where
//   n = wcol*16 + (lane&15),  k = kstep*32 + (lane>>4)*16 + i   (contiguous k order, same as A fragment)
__global__ void k_convert_w(const float* __restrict__ W1, const float* __restrict__ W2,
                            const float* __restrict__ W3, __bf16* __restrict__ wb) {
    int idx = blockIdx.x * blockDim.x + threadIdx.x;
    if (idx >= 3 * 16384) return;
    int l    = idx >> 14;
    int r    = idx & 16383;
    int wcol = r >> 11;
    int ks   = (r >> 9) & 3;
    int lane = (r >> 4) & 31;
    int i    = r & 15;
    int n    = wcol * 16 + (lane & 15);
    int k    = ks * 32 + (lane >> 4) * 16 + i;
    const float* W = (l == 0) ? W1 : ((l == 1) ? W2 : W3);
    wb[idx] = f2bf(W[k * D + n]);
}

// deg[dst] += 1 per real edge (self-loop +1 folded into k_dinv)
__global__ void k_deg(const int* __restrict__ dst, float* __restrict__ deg) {
    int e = blockIdx.x * blockDim.x + threadIdx.x;
    if (e < N_EDGES) atomicAdd(&deg[dst[e]], 1.0f);
}

// in-place: deg -> dinv = rsqrt(max(deg+1, 1))
__global__ void k_dinv(float* __restrict__ deg) {
    int i = blockIdx.x * blockDim.x + threadIdx.x;
    if (i < N_NODES) deg[i] = rsqrtf(fmaxf(deg[i] + 1.0f, 1.0f));
}

// ---------------- WMMA GEMM: out[N,128] = actfn(A)[N,128] @ W[128,128] ----------------
// actfn(v) = relu(v + bias[col]) when bias != nullptr (fuses previous layer's bias+ReLU),
// else identity. One block = 8 waves = one 16-row strip; wave w -> cols [16w,16w+16).
__global__ void __launch_bounds__(256) k_gemm(const float* __restrict__ A,
                                              const __bf16* __restrict__ wb,
                                              const float* __restrict__ bias,
                                              float* __restrict__ out) {
    __shared__ __align__(16) __bf16 As[16][136];   // padded stride: 272B -> conflict-light

    const int row0 = blockIdx.x * 16;
    const int tid  = threadIdx.x;

    // Cooperative stage: 16x128 fp32 strip -> bf16 LDS, with fused bias+ReLU
#pragma unroll
    for (int j = 0; j < 8; ++j) {
        int idx = tid + j * 256;          // 0..2047
        int rr  = idx >> 7;
        int col = idx & 127;
        float v = A[(row0 + rr) * D + col];
        if (bias) v = fmaxf(v + bias[col], 0.0f);
        As[rr][col] = f2bf(v);
    }
    __syncthreads();

    const int w    = tid >> 5;
    const int lane = tid & 31;
    const int half = lane >> 4;
    const int m    = lane & 15;

    v8f c = {};
#pragma unroll
    for (int ks = 0; ks < 4; ++ks) {
        // A fragment: contiguous k = 16*half + s per lane (matches B packing)
        const __bf16* arow = &As[m][ks * 32 + half * 16];
        v4u alo = *(const v4u*)(arow);
        v4u ahi = *(const v4u*)(arow + 8);
        v16bf a = __builtin_bit_cast(v16bf,
                   __builtin_shufflevector(alo, ahi, 0, 1, 2, 3, 4, 5, 6, 7));
        // B fragment: pre-swizzled -> one contiguous 32B load per lane
        v8u braw = *(const v8u*)(wb + (((w * 4 + ks) * 32 + lane) << 4));
        v16bf b  = __builtin_bit_cast(v16bf, braw);
        c = __builtin_amdgcn_wmma_f32_16x16x32_bf16(false, a, false, b,
                                                    (short)0, c, false, false);
    }

    // C/D layout: VGPR r -> M = r + half*8, N = lane&15
    float* orow = out + (size_t)row0 * D + w * 16 + (lane & 15);
#pragma unroll
    for (int r = 0; r < 8; ++r) orow[(size_t)(r + half * 8) * D] = c[r];
}

// acc = t * dinv^2  (self-loop term; also serves as the zero-init of the accumulator)
__global__ void k_selfloop(const float* __restrict__ t, const float* __restrict__ dinv,
                           float* __restrict__ acc) {
    int idx = blockIdx.x * blockDim.x + threadIdx.x;
    if (idx < N_NODES * D) {
        int i = idx >> 7;
        float di = dinv[i];
        acc[idx] = t[idx] * di * di;
    }
}

// One wave per edge: float4 gather of t[src], scaled atomic scatter into acc[dst]
__global__ void __launch_bounds__(256) k_scatter(const float* __restrict__ t,
                                                 const float* __restrict__ dinv,
                                                 const int* __restrict__ src,
                                                 const int* __restrict__ dst,
                                                 float* __restrict__ acc) {
    int e = blockIdx.x * 8 + (threadIdx.x >> 5);
    if (e >= N_EDGES) return;
    int lane = threadIdx.x & 31;
    int s = src[e], d = dst[e];
    float wgt = dinv[s] * dinv[d];
    const float4* tp = (const float4*)(t + (size_t)s * D);
    float4 v = tp[lane];
    float* ap = acc + (size_t)d * D + lane * 4;
    atomicAdd(ap + 0, v.x * wgt);
    atomicAdd(ap + 1, v.y * wgt);
    atomicAdd(ap + 2, v.z * wgt);
    atomicAdd(ap + 3, v.w * wgt);
}

// Segment-sum pooling with fused bias+ReLU of layer 3
__global__ void k_pool(const float* __restrict__ acc, const float* __restrict__ b3,
                       const int* __restrict__ batch, float* __restrict__ pooled,
                       float* __restrict__ cnt) {
    int idx = blockIdx.x * blockDim.x + threadIdx.x;
    if (idx >= N_NODES * D) return;
    int i = idx >> 7, f = idx & 127;
    int g = batch[i];
    float v = fmaxf(acc[idx] + b3[f], 0.0f);
    atomicAdd(&pooled[g * D + f], v);
    if (f == 0) atomicAdd(&cnt[g], 1.0f);
}

// One wave per graph: mean, 128-dot with Wfc, sigmoid
__global__ void __launch_bounds__(256) k_final(const float* __restrict__ pooled,
                                               const float* __restrict__ cnt,
                                               const float* __restrict__ Wfc,
                                               const float* __restrict__ bfc,
                                               float* __restrict__ out) {
    int g = blockIdx.x * 8 + (threadIdx.x >> 5);
    if (g >= N_GRAPHS) return;
    int lane = threadIdx.x & 31;
    const float4* pp = (const float4*)(pooled + (size_t)g * D);
    const float4* wp = (const float4*)Wfc;
    float4 p = pp[lane], w = wp[lane];
    float s = p.x * w.x + p.y * w.y + p.z * w.z + p.w * w.w;
#pragma unroll
    for (int off = 16; off >= 1; off >>= 1) s += __shfl_xor(s, off, 32);
    if (lane == 0) {
        float c = fmaxf(cnt[g], 1.0f);
        float z = s / c + bfc[0];
        out[g] = 1.0f / (1.0f + __expf(-z));
    }
}

extern "C" void kernel_launch(void* const* d_in, const int* in_sizes, int n_in,
                              void* d_out, int out_size, void* d_ws, size_t ws_size,
                              hipStream_t stream) {
    (void)in_sizes; (void)n_in; (void)out_size; (void)ws_size;

    const float* x    = (const float*)d_in[0];
    const int*   ei   = (const int*)d_in[1];   // (2, E): row0 = src, row1 = dst
    const int*   src  = ei;
    const int*   dstp = ei + N_EDGES;
    const int*   batch = (const int*)d_in[2];
    const float* W1 = (const float*)d_in[3];
    const float* b1 = (const float*)d_in[4];
    const float* W2 = (const float*)d_in[5];
    const float* b2 = (const float*)d_in[6];
    const float* W3 = (const float*)d_in[7];
    const float* b3 = (const float*)d_in[8];
    const float* Wfc = (const float*)d_in[9];
    const float* bfc = (const float*)d_in[10];
    float* out = (float*)d_out;

    // ---- workspace layout ----
    char* ws = (char*)d_ws;
    size_t off = 0;
    __bf16* wb   = (__bf16*)(ws + off); off += (size_t)3 * 16384 * 2;        // 96 KB
    float*  buf0 = (float*)(ws + off);  off += (size_t)N_NODES * D * 4;      // 51.2 MB
    float*  buf1 = (float*)(ws + off);  off += (size_t)N_NODES * D * 4;      // 51.2 MB
    float*  deg  = (float*)(ws + off);  off += 400128;                       // N*4 + pad
    float*  pooled = (float*)(ws + off); off += (size_t)N_GRAPHS * D * 4;    // 256 KB
    float*  cnt  = (float*)(ws + off);  off += (size_t)N_GRAPHS * 4;

    // zero deg | pooled | cnt (contiguous region)
    int zero_n = (400128 + N_GRAPHS * D * 4 + N_GRAPHS * 4) / 4;
    k_zero<<<(zero_n + 255) / 256, 256, 0, stream>>>(deg, zero_n);

    k_convert_w<<<192, 256, 0, stream>>>(W1, W2, W3, wb);
    k_deg<<<N_EDGES / 256, 256, 0, stream>>>(dstp, deg);
    k_dinv<<<(N_NODES + 255) / 256, 256, 0, stream>>>(deg);

    const int gGemm = N_NODES / 16;          // 6250
    const int gElem = (N_NODES * D) / 256;   // 50000
    const int gEdge = N_EDGES / 8;           // 400000

    // layer 1
    k_gemm<<<gGemm, 256, 0, stream>>>(x, wb, nullptr, buf0);
    k_selfloop<<<gElem, 256, 0, stream>>>(buf0, deg, buf1);
    k_scatter<<<gEdge, 256, 0, stream>>>(buf0, deg, src, dstp, buf1);
    // layer 2 (bias1+ReLU fused into A staging)
    k_gemm<<<gGemm, 256, 0, stream>>>(buf1, wb + 16384, b1, buf0);
    k_selfloop<<<gElem, 256, 0, stream>>>(buf0, deg, buf1);
    k_scatter<<<gEdge, 256, 0, stream>>>(buf0, deg, src, dstp, buf1);
    // layer 3
    k_gemm<<<gGemm, 256, 0, stream>>>(buf1, wb + 2 * 16384, b2, buf0);
    k_selfloop<<<gElem, 256, 0, stream>>>(buf0, deg, buf1);
    k_scatter<<<gEdge, 256, 0, stream>>>(buf0, deg, src, dstp, buf1);
    // pooling (bias3+ReLU fused) + head
    k_pool<<<gElem, 256, 0, stream>>>(buf1, b3, batch, pooled, cnt);
    k_final<<<N_GRAPHS / 8, 256, 0, stream>>>(pooled, cnt, Wfc, bfc, out);
}
